// GraphNodeRefinement_76347338654280
// MI455X (gfx1250) — compile-verified, matrix operand order checked by reading, not run on hardware
//
#include <hip/hip_runtime.h>
#include <hip/hip_bf16.h>

typedef __attribute__((ext_vector_type(16))) _Float16 v16h;
typedef __attribute__((ext_vector_type(8)))  float    v8f;

#define NN     50000
#define EE     800000
#define E2     (EE + NN)      // edges + self loops
#define HDIM   128
#define HEADS  4
#define CDIM   32
#define LAYERS 3
#define NGRAPH 64
#define NCLS   10

// ---------- helpers: order-preserving float<->uint map for atomic max ----------
__device__ __forceinline__ unsigned om(float f) {
    unsigned u = __float_as_uint(f);
    return (u & 0x80000000u) ? ~u : (u | 0x80000000u);
}
__device__ __forceinline__ float od(unsigned x) {
    unsigned u = (x & 0x80000000u) ? (x ^ 0x80000000u) : ~x;
    return __uint_as_float(u);
}

// ---------- weight packing into per-lane WMMA B-fragment layout ----------
// Target layout: wp[mtx][ ((chunk*8 + tile)*32 + lane)*16 + j ]
//   chunk: K-block of 32 (0..3), tile: N-block of 16 (0..7)
//   lane<16: kb=0 (K 0..7,16..23), lane>=16: kb=8 (K 8..15,24..31)
//   half j: K = chunk*32 + kb + (j<8 ? j : j+8), col = tile*16 + (lane&15)
__global__ void pack_kernel(const float* __restrict__ Wn,
                            const float* __restrict__ Wg,
                            _Float16* __restrict__ wp) {
    int t = blockIdx.x * blockDim.x + threadIdx.x;
    if (t >= 4 * 4 * 8 * 32 * 16) return;
    int j     =  t        & 15;
    int lane  = (t >> 4)  & 31;
    int tile  = (t >> 9)  & 7;
    int chunk = (t >> 12) & 3;
    int mtx   =  t >> 14;
    const float* W = (mtx == 0) ? Wn : (Wg + (size_t)(mtx - 1) * HDIM * HDIM);
    int kb  = (lane < 16) ? 0 : 8;
    int k   = chunk * 32 + kb + (j < 8 ? j : j + 8);
    int col = tile * 16 + (lane & 15);
    wp[t] = (_Float16)W[(size_t)k * HDIM + col];
}

// ---------- WMMA GEMM: D[rows x 128] = A[rows x 128] @ Bpacked (+bias +emb) ----------
// one wave computes a 16x128 row-block: 8 N-tiles x 4 K-chunks = 32 WMMAs
__global__ __launch_bounds__(256)
void gemm_wmma(const float* __restrict__ A, const _Float16* __restrict__ Bp,
               float* __restrict__ D, const float* __restrict__ bias,
               const float* __restrict__ emb, const int* __restrict__ types,
               int nrows) {
    int wave  = threadIdx.x >> 5;
    int lane  = threadIdx.x & 31;
    int tileM = blockIdx.x * 8 + wave;
    if (tileM * 16 >= nrows) return;
    int row0 = tileM * 16;
    int arow = row0 + (lane & 15);
    int kb   = (lane < 16) ? 0 : 8;

    v8f acc[8];
#pragma unroll
    for (int t = 0; t < 8; t++) acc[t] = (v8f){0.f,0.f,0.f,0.f,0.f,0.f,0.f,0.f};

#pragma unroll
    for (int chunk = 0; chunk < 4; ++chunk) {
        const float* ap = A + (size_t)arow * HDIM + chunk * 32 + kb;
        v16h afrag;
#pragma unroll
        for (int j = 0; j < 8; j++) afrag[j]     = (_Float16)ap[j];
#pragma unroll
        for (int j = 0; j < 8; j++) afrag[j + 8] = (_Float16)ap[16 + j];
#pragma unroll
        for (int t = 0; t < 8; t++) {
            v16h bfrag = *(const v16h*)(Bp + ((((size_t)chunk * 8 + t) * 32) + lane) * 16);
            acc[t] = __builtin_amdgcn_wmma_f32_16x16x32_f16(
                false, afrag, false, bfrag, (short)0, acc[t], false, false);
        }
    }

    // epilogue: VGPR r -> row = row0 + r + (lane<16?0:8); col = tile*16 + (lane&15)
    int col0  = lane & 15;
    int rbase = row0 + ((lane < 16) ? 0 : 8);
#pragma unroll
    for (int r = 0; r < 8; r++) {
        int row = rbase + r;
        const float* erow = emb ? (emb + (size_t)types[row] * HDIM) : nullptr;
#pragma unroll
        for (int t = 0; t < 8; t++) {
            int col = t * 16 + col0;
            float v = acc[t][r];
            if (bias) v += bias[col];
            if (erow) v += erow[col];
            D[(size_t)row * HDIM + col] = v;
        }
    }
}

// ---------- per-node attention scores ----------
__global__ void score_kernel(const float* __restrict__ xh,
                             const float* __restrict__ asrc,
                             const float* __restrict__ adst,
                             float* __restrict__ ssrc, float* __restrict__ sdst) {
    int i = blockIdx.x * blockDim.x + threadIdx.x;
    if (i >= NN * HEADS) return;
    int node = i >> 2, head = i & 3;
    const float* xr = xh + (size_t)node * HDIM + head * CDIM;
    const float* as = asrc + head * CDIM;
    const float* ad = adst + head * CDIM;
    float s0 = 0.f, s1 = 0.f;
#pragma unroll 8
    for (int c = 0; c < CDIM; c++) { float v = xr[c]; s0 += v * as[c]; s1 += v * ad[c]; }
    ssrc[i] = s0; sdst[i] = s1;
}

// ---------- per-layer init: m=-inf(mapped), z=0, acc=0 ----------
__global__ void layer_init(unsigned* __restrict__ m, float* __restrict__ z,
                           float* __restrict__ acc) {
    int i = blockIdx.x * blockDim.x + threadIdx.x;
    if (i < NN * HDIM) acc[i] = 0.f;
    if (i < NN * HEADS) { m[i] = 0x007FFFFFu; z[i] = 0.f; }   // om(-inf)
}

// ---------- edge pass 1: e = leaky_relu(s_src[src]+s_dst[dst]); segment max ----------
__global__ void edge_score(const int* __restrict__ ei,
                           const float* __restrict__ ssrc, const float* __restrict__ sdst,
                           float* __restrict__ ebuf, unsigned* __restrict__ mbuf) {
    int e = blockIdx.x * blockDim.x + threadIdx.x;
    if (e >= E2) return;
    int s, d;
    if (e < EE) { s = ei[e]; d = ei[EE + e]; } else { s = d = e - EE; }
#pragma unroll
    for (int h = 0; h < HEADS; h++) {
        float v = ssrc[s * HEADS + h] + sdst[d * HEADS + h];
        v = v > 0.f ? v : 0.2f * v;
        ebuf[(size_t)e * HEADS + h] = v;
        atomicMax(&mbuf[d * HEADS + h], om(v));
    }
}

// ---------- edge pass 2: ee = exp(e - m[dst]); segment sum z ----------
__global__ void edge_exp(const int* __restrict__ ei,
                         const unsigned* __restrict__ mbuf,
                         float* __restrict__ ebuf, float* __restrict__ zbuf) {
    int e = blockIdx.x * blockDim.x + threadIdx.x;
    if (e >= E2) return;
    int d = (e < EE) ? ei[EE + e] : (e - EE);
#pragma unroll
    for (int h = 0; h < HEADS; h++) {
        float m  = od(mbuf[d * HEADS + h]);
        float ee = expf(ebuf[(size_t)e * HEADS + h] - m);
        ebuf[(size_t)e * HEADS + h] = ee;
        atomicAdd(&zbuf[d * HEADS + h], ee);
    }
}

// ---------- edge pass 3: message scatter (wave per edge, lane = feature) ----------
__global__ __launch_bounds__(256)
void edge_msg(const int* __restrict__ ei, const float* __restrict__ xh,
              const float* __restrict__ ebuf, const float* __restrict__ zbuf,
              float* __restrict__ acc) {
    int wid  = (int)((blockIdx.x * (size_t)blockDim.x + threadIdx.x) >> 5);
    int lane = threadIdx.x & 31;
    if (wid >= E2) return;
    int s, d;
    if (wid < EE) { s = ei[wid]; d = ei[EE + wid]; } else { s = d = wid - EE; }
#pragma unroll
    for (int h = 0; h < HEADS; h++) {
        float alpha = ebuf[(size_t)wid * HEADS + h] / (zbuf[d * HEADS + h] + 1e-16f);
        float v = xh[(size_t)s * HDIM + h * CDIM + lane] * alpha;
        atomicAdd(&acc[(size_t)d * HDIM + h * CDIM + lane], v);
    }
}

// ---------- bias + layernorm + residual + relu (wave per node) ----------
__global__ __launch_bounds__(256)
void ln_kernel(const float* __restrict__ acc, const float* __restrict__ bgat,
               const float* __restrict__ hprev, const float* __restrict__ g,
               const float* __restrict__ bln, float* __restrict__ hout) {
    int node = (int)((blockIdx.x * (size_t)blockDim.x + threadIdx.x) >> 5);
    int lane = threadIdx.x & 31;
    if (node >= NN) return;
    float v[4];
    float sum = 0.f;
#pragma unroll
    for (int i = 0; i < 4; i++) {
        v[i] = acc[(size_t)node * HDIM + i * 32 + lane] + bgat[i * 32 + lane];
        sum += v[i];
    }
#pragma unroll
    for (int o = 16; o > 0; o >>= 1) sum += __shfl_xor(sum, o, 32);
    float mu = sum * (1.f / HDIM);
    float var = 0.f;
#pragma unroll
    for (int i = 0; i < 4; i++) { float dd = v[i] - mu; var += dd * dd; }
#pragma unroll
    for (int o = 16; o > 0; o >>= 1) var += __shfl_xor(var, o, 32);
    float rs = rsqrtf(var * (1.f / HDIM) + 1e-5f);
#pragma unroll
    for (int i = 0; i < 4; i++) {
        int f = i * 32 + lane;
        float ln = (v[i] - mu) * rs * g[f] + bln[f];
        float r  = ln + hprev[(size_t)node * HDIM + f];
        hout[(size_t)node * HDIM + f] = r > 0.f ? r : 0.f;
    }
}

// ---------- pooling ----------
__global__ void zero_pool(float* __restrict__ sums, float* __restrict__ cnt) {
    int i = blockIdx.x * blockDim.x + threadIdx.x;
    if (i < NGRAPH * HDIM) sums[i] = 0.f;
    if (i < NGRAPH) cnt[i] = 0.f;
}
__global__ __launch_bounds__(256)
void pool_kernel(const float* __restrict__ h, const int* __restrict__ batch,
                 float* __restrict__ sums, float* __restrict__ cnt) {
    int node = (int)((blockIdx.x * (size_t)blockDim.x + threadIdx.x) >> 5);
    int lane = threadIdx.x & 31;
    if (node >= NN) return;
    int g = batch[node];
#pragma unroll
    for (int i = 0; i < 4; i++)
        atomicAdd(&sums[(size_t)g * HDIM + i * 32 + lane],
                  h[(size_t)node * HDIM + i * 32 + lane]);
    if (lane == 0) atomicAdd(&cnt[g], 1.0f);
}

// ---------- tiny MLP head ----------
__global__ void hg_finalize(const float* __restrict__ sums, const float* __restrict__ cnt,
                            float* __restrict__ hg) {
    int i = blockIdx.x * blockDim.x + threadIdx.x;
    if (i >= NGRAPH * HDIM) return;
    float c = cnt[i >> 7]; c = c > 1.f ? c : 1.f;
    hg[i] = sums[i] / c;
}
__global__ void mlp1(const float* __restrict__ hg, const float* __restrict__ W1,
                     const float* __restrict__ b1, float* __restrict__ z1) {
    int i = blockIdx.x * blockDim.x + threadIdx.x;
    if (i >= NGRAPH * 256) return;
    int g = i >> 8, o = i & 255;
    float s = b1[o];
    const float* hr = hg + (size_t)g * HDIM;
    for (int k = 0; k < HDIM; k++) s += hr[k] * W1[(size_t)k * 256 + o];
    z1[i] = s > 0.f ? s : 0.f;
}
__global__ void mlp2(const float* __restrict__ z1, const float* __restrict__ W2,
                     const float* __restrict__ b2, float* __restrict__ z2) {
    int i = blockIdx.x * blockDim.x + threadIdx.x;
    if (i >= NGRAPH * HDIM) return;
    int g = i >> 7, o = i & 127;
    float s = b2[o];
    const float* zr = z1 + (size_t)g * 256;
    for (int k = 0; k < 256; k++) s += zr[k] * W2[(size_t)k * HDIM + o];
    z2[i] = s > 0.f ? s : 0.f;
}
__global__ void mlp3(const float* __restrict__ z2, const float* __restrict__ W3,
                     const float* __restrict__ b3, float* __restrict__ out) {
    int i = blockIdx.x * blockDim.x + threadIdx.x;
    if (i >= NGRAPH * NCLS) return;
    int g = i / NCLS, o = i % NCLS;
    float s = b3[o];
    const float* zr = z2 + (size_t)g * HDIM;
    for (int k = 0; k < HDIM; k++) s += zr[k] * W3[(size_t)k * NCLS + o];
    out[i] = s;
}

extern "C" void kernel_launch(void* const* d_in, const int* in_sizes, int n_in,
                              void* d_out, int out_size, void* d_ws, size_t ws_size,
                              hipStream_t stream) {
    const float* x        = (const float*)d_in[0];
    const int*   ei       = (const int*)  d_in[1];
    const int*   ntypes   = (const int*)  d_in[2];
    const int*   batch    = (const int*)  d_in[3];
    const float* W_node   = (const float*)d_in[4];
    const float* b_node   = (const float*)d_in[5];
    const float* type_emb = (const float*)d_in[6];
    const float* W_gat    = (const float*)d_in[7];
    const float* a_src    = (const float*)d_in[8];
    const float* a_dst    = (const float*)d_in[9];
    const float* b_gat    = (const float*)d_in[10];
    const float* ln_g     = (const float*)d_in[11];
    const float* ln_b     = (const float*)d_in[12];
    const float* W1       = (const float*)d_in[13];
    const float* b1       = (const float*)d_in[14];
    const float* W2       = (const float*)d_in[15];
    const float* b2       = (const float*)d_in[16];
    const float* W3       = (const float*)d_in[17];
    const float* b3       = (const float*)d_in[18];
    float* out = (float*)d_out;

    // workspace layout
    char* w = (char*)d_ws; size_t off = 0;
    auto alloc = [&](size_t bytes) -> void* {
        void* p = w + off; off += (bytes + 255) & ~(size_t)255; return p;
    };
    float*    P0    = (float*)   alloc((size_t)NN * HDIM * 4);
    float*    P1    = (float*)   alloc((size_t)NN * HDIM * 4);
    float*    P2    = (float*)   alloc((size_t)NN * HDIM * 4);
    float*    ebuf  = (float*)   alloc((size_t)E2 * HEADS * 4);
    float*    ssrc  = (float*)   alloc((size_t)NN * HEADS * 4);
    float*    sdst  = (float*)   alloc((size_t)NN * HEADS * 4);
    unsigned* mbuf  = (unsigned*)alloc((size_t)NN * HEADS * 4);
    float*    zbuf  = (float*)   alloc((size_t)NN * HEADS * 4);
    _Float16* wpack = (_Float16*)alloc((size_t)4 * HDIM * HDIM * 2);
    float*    hgsum = (float*)   alloc((size_t)NGRAPH * HDIM * 4);
    float*    cnt   = (float*)   alloc((size_t)NGRAPH * 4);
    float*    hg    = (float*)   alloc((size_t)NGRAPH * HDIM * 4);
    float*    z1    = (float*)   alloc((size_t)NGRAPH * 256 * 4);
    float*    z2    = (float*)   alloc((size_t)NGRAPH * HDIM * 4);
    (void)ws_size; (void)n_in; (void)in_sizes; (void)out_size;

    const int B = 256;

    // pack all weight matrices into WMMA B-fragment layout
    pack_kernel<<<(4 * HDIM * HDIM + B - 1) / B, B, 0, stream>>>(W_node, W_gat, wpack);

    // h = x @ W_node + b_node + type_emb[node_types]
    const int ntiles = NN / 16;                 // 3125
    const int gemmGrid = (ntiles + 7) / 8;      // 8 waves per block
    gemm_wmma<<<gemmGrid, B, 0, stream>>>(x, wpack, P0, b_node, type_emb, ntypes, NN);

    float* hcur = P0;   // current features
    float* xh   = P1;   // h @ W per layer
    float* accb = P2;   // aggregation target

    const int edgeGrid  = (E2 + B - 1) / B;
    const int waveEdge  = (E2 * 32 + B - 1) / B;
    const int waveNode  = (NN * 32 + B - 1) / B;
    const int initGrid  = (NN * HDIM + B - 1) / B;

    for (int l = 0; l < LAYERS; l++) {
        gemm_wmma<<<gemmGrid, B, 0, stream>>>(
            hcur, wpack + (size_t)(1 + l) * HDIM * HDIM, xh,
            nullptr, nullptr, nullptr, NN);
        score_kernel<<<(NN * HEADS + B - 1) / B, B, 0, stream>>>(
            xh, a_src + (size_t)l * HEADS * CDIM, a_dst + (size_t)l * HEADS * CDIM,
            ssrc, sdst);
        layer_init<<<initGrid, B, 0, stream>>>(mbuf, zbuf, accb);
        edge_score<<<edgeGrid, B, 0, stream>>>(ei, ssrc, sdst, ebuf, mbuf);
        edge_exp<<<edgeGrid, B, 0, stream>>>(ei, mbuf, ebuf, zbuf);
        edge_msg<<<waveEdge, B, 0, stream>>>(ei, xh, ebuf, zbuf, accb);
        // in-place LN/residual into accb, then rotate: accb becomes new h
        ln_kernel<<<waveNode, B, 0, stream>>>(
            accb, b_gat + (size_t)l * HDIM, hcur,
            ln_g + (size_t)l * HDIM, ln_b + (size_t)l * HDIM, accb);
        float* tmp = hcur; hcur = accb; accb = tmp;
    }

    // global mean pool + MLP head
    zero_pool<<<(NGRAPH * HDIM + B - 1) / B, B, 0, stream>>>(hgsum, cnt);
    pool_kernel<<<waveNode, B, 0, stream>>>(hcur, batch, hgsum, cnt);
    hg_finalize<<<(NGRAPH * HDIM + B - 1) / B, B, 0, stream>>>(hgsum, cnt, hg);
    mlp1<<<(NGRAPH * 256 + B - 1) / B, B, 0, stream>>>(hg, W1, b1, z1);
    mlp2<<<(NGRAPH * HDIM + B - 1) / B, B, 0, stream>>>(z1, W2, b2, z2);
    mlp3<<<(NGRAPH * NCLS + B - 1) / B, B, 0, stream>>>(z2, W3, b3, out);
}
